// NERBertWithCRF_25735444038438
// MI455X (gfx1250) — matrix operand order repaired and checked by reading, not compiled, not run on hardware
//
#include <hip/hip_runtime.h>
#include <hip/hip_bf16.h>
#include <math.h>

// ---------------------------------------------------------------------------
// Model dims (compile-time constants matching the reference)
// ---------------------------------------------------------------------------
#define BB   32
#define LL   128
#define DMOD 768
#define NLAY 12
#define NHEA 12
#define DHEA 64
#define FFD  3072
#define KTAG 9
#define TT_  127
#define MROW 4096        // B*L

typedef __bf16 bf16;
typedef __attribute__((ext_vector_type(8)))  __bf16 v8bf;
typedef __attribute__((ext_vector_type(16))) __bf16 v16bf;
typedef __attribute__((ext_vector_type(8)))  float  v8f;

__device__ __forceinline__ v16bf make_frag(const bf16* lo, const bf16* hi) {
    v8bf a = *(const v8bf*)lo;
    v8bf b = *(const v8bf*)hi;
    return __builtin_shufflevector(a, b, 0,1,2,3,4,5,6,7,8,9,10,11,12,13,14,15);
}

__device__ __forceinline__ v8f wmma_bf16(v16bf a, v16bf b, v8f c) {
    return __builtin_amdgcn_wmma_f32_16x16x32_bf16(false, a, false, b, (short)0, c, false, false);
}

// async copy of 16 bytes global -> LDS (CDNA5 GLOBAL_LOAD_ASYNC_TO_LDS_B128,
// tracked by ASYNCcnt). ldsaddr = low 32 bits of generic pointer to LDS.
__device__ __forceinline__ void async_copy_b128(unsigned lds_addr, const void* gptr) {
    asm volatile("global_load_async_to_lds_b128 %0, %1, off"
                 :: "v"(lds_addr), "v"(gptr) : "memory");
}
__device__ __forceinline__ void wait_async0() {
    asm volatile("s_wait_asynccnt 0x0" ::: "memory");
}

// ---------------------------------------------------------------------------
// Weight convert + transpose: W[Kd][N] f32  ->  WT[N][Kd] bf16 (per layer z)
// 32x32 LDS tiled transpose, both global accesses coalesced.
// ---------------------------------------------------------------------------
__global__ __launch_bounds__(256) void wcvt_kernel(
    const float* __restrict__ W, bf16* __restrict__ WT, int Kd, int N)
{
    const size_t lofs = (size_t)blockIdx.z * Kd * N;
    const float* Wl = W + lofs;
    bf16* WTl = WT + lofs;
    __shared__ float t[32][33];
    const int n0 = blockIdx.x * 32;
    const int k0 = blockIdx.y * 32;
    const int c  = threadIdx.x & 31;
    const int r4 = threadIdx.x >> 5;         // 0..7
    for (int rr = 0; rr < 32; rr += 8)
        t[rr + r4][c] = Wl[(size_t)(k0 + rr + r4) * N + n0 + c];
    __syncthreads();
    for (int rr = 0; rr < 32; rr += 8)
        WTl[(size_t)(n0 + rr + r4) * Kd + k0 + c] = (bf16)t[c][rr + r4];
}

// ---------------------------------------------------------------------------
// Embedding gather + LayerNorm  ->  X[4096,768] f32
// ---------------------------------------------------------------------------
__global__ __launch_bounds__(256) void embed_ln_kernel(
    const int* __restrict__ ids, const int* __restrict__ tti,
    const float* __restrict__ wemb, const float* __restrict__ pemb,
    const float* __restrict__ temb, const float* __restrict__ g,
    const float* __restrict__ bta, float* __restrict__ X)
{
    const int row = blockIdx.x;            // 0..4095
    const int l   = row & (LL - 1);
    const int id  = ids[row];
    const int tt  = tti[row];
    __shared__ float buf[DMOD];
    __shared__ float r1[256], r2[256];
    const int tid = threadIdx.x;
    float lsum = 0.f, lsq = 0.f;
    for (int d = tid; d < DMOD; d += 256) {
        float x = wemb[(size_t)id * DMOD + d] + pemb[l * DMOD + d] + temb[tt * DMOD + d];
        buf[d] = x; lsum += x; lsq += x * x;
    }
    r1[tid] = lsum; r2[tid] = lsq;
    __syncthreads();
    for (int s = 128; s > 0; s >>= 1) {
        if (tid < s) { r1[tid] += r1[tid + s]; r2[tid] += r2[tid + s]; }
        __syncthreads();
    }
    const float mean = r1[0] * (1.0f / DMOD);
    const float var  = r2[0] * (1.0f / DMOD) - mean * mean;
    const float rstd = rsqrtf(var + 1e-12f);
    for (int d = tid; d < DMOD; d += 256)
        X[(size_t)row * DMOD + d] = (buf[d] - mean) * rstd * g[d] + bta[d];
}

// ---------------------------------------------------------------------------
// LayerNorm kernel: Out = LN(In) * g + b   (row-wise over 768)
// ---------------------------------------------------------------------------
__global__ __launch_bounds__(256) void ln_kernel(
    const float* __restrict__ In, const float* __restrict__ g,
    const float* __restrict__ bta, float* __restrict__ Out)
{
    const int row = blockIdx.x;
    __shared__ float buf[DMOD];
    __shared__ float r1[256], r2[256];
    const int tid = threadIdx.x;
    float lsum = 0.f, lsq = 0.f;
    for (int d = tid; d < DMOD; d += 256) {
        float x = In[(size_t)row * DMOD + d];
        buf[d] = x; lsum += x; lsq += x * x;
    }
    r1[tid] = lsum; r2[tid] = lsq;
    __syncthreads();
    for (int s = 128; s > 0; s >>= 1) {
        if (tid < s) { r1[tid] += r1[tid + s]; r2[tid] += r2[tid + s]; }
        __syncthreads();
    }
    const float mean = r1[0] * (1.0f / DMOD);
    const float var  = r2[0] * (1.0f / DMOD) - mean * mean;
    const float rstd = rsqrtf(var + 1e-12f);
    for (int d = tid; d < DMOD; d += 256)
        Out[(size_t)row * DMOD + d] = (buf[d] - mean) * rstd * g[d] + bta[d];
}

// ---------------------------------------------------------------------------
// WMMA GEMM: C[M,N] = epi(A[M,Kd] * W[Kd,N] + bias [+ residual])
//   A  : f32 activations, converted to bf16 through VGPRs into LDS
//   WT : pre-transposed bf16 weights [N][Kd]; B tiles pulled with
//        GLOBAL_LOAD_ASYNC_TO_LDS_B128 (no VGPR round trip, ASYNCcnt tracked)
// block = 256 threads = 8 waves; tile 64x128; BK = 32
// EPI: 0 = bias, 1 = bias+exact GELU, 2 = bias+residual
// ---------------------------------------------------------------------------
#define GBM 64
#define GBN 128
#define GBK 32
#define ASTR 40   // halves, row stride (80B, 16B multiple)
#define BSTR 40

template<int EPI>
__global__ __launch_bounds__(256) void gemm_bf16_kernel(
    const float* __restrict__ A, const bf16* __restrict__ WT,
    const float* __restrict__ bias, const float* __restrict__ Res,
    float* __restrict__ C, int Kd, int N)
{
    __shared__ __align__(16) bf16 As[GBM * ASTR];
    __shared__ __align__(16) bf16 Bs[GBN * BSTR];

    const int tid   = threadIdx.x;
    const int m0    = blockIdx.y * GBM;
    const int n0    = blockIdx.x * GBN;
    const int w     = tid >> 5;
    const int lane  = tid & 31;
    const int wr    = w >> 1;
    const int wc    = w & 1;
    const int mlane = lane & 15;
    const int khalf = lane >> 4;

    v8f acc[4] = {};

    for (int k0 = 0; k0 < Kd; k0 += GBK) {
        // ---- async-stage B tile: WT rows n0..n0+127, 64B (32 bf16) each ----
        for (int i = tid; i < GBN * 4; i += 256) {
            int n = i >> 2, c4 = i & 3;
            const bf16* g = WT + (size_t)(n0 + n) * Kd + k0 + c4 * 8;
            unsigned laddr = (unsigned)(uintptr_t)(&Bs[n * BSTR + c4 * 8]);
            async_copy_b128(laddr, g);
        }
        // ---- stage A tile 64x32 (f32 -> bf16) overlapped with async B ----
        for (int i = tid; i < GBM * GBK; i += 256) {
            int r = i >> 5, k = i & 31;
            As[r * ASTR + k] = (bf16)A[(size_t)(m0 + r) * Kd + k0 + k];
        }
        if (k0 + GBK < Kd)   // hint next A k-tile into cache
            __builtin_prefetch(&A[(size_t)(m0 + (tid & 63)) * Kd + k0 + GBK], 0, 1);
        wait_async0();
        __syncthreads();

        const int arow = wr * 16 + mlane;
        v16bf afrag = make_frag(&As[arow * ASTR + khalf * 8],
                                &As[arow * ASTR + 16 + khalf * 8]);
#pragma unroll
        for (int t = 0; t < 4; t++) {
            int bn = wc * 64 + t * 16 + mlane;
            v16bf bfrag = make_frag(&Bs[bn * BSTR + khalf * 8],
                                    &Bs[bn * BSTR + 16 + khalf * 8]);
            acc[t] = wmma_bf16(afrag, bfrag, acc[t]);
        }
        __syncthreads();
    }

#pragma unroll
    for (int t = 0; t < 4; t++) {
        const int n  = n0 + wc * 64 + t * 16 + mlane;
        const float bv = bias[n];
#pragma unroll
        for (int r = 0; r < 8; r++) {
            const int m = m0 + wr * 16 + khalf * 8 + r;
            float val = acc[t][r] + bv;
            if (EPI == 1) val = 0.5f * val * (1.0f + erff(val * 0.70710678118654752f));
            if (EPI == 2) val += Res[(size_t)m * N + n];
            C[(size_t)m * N + n] = val;
        }
    }
}

// ---------------------------------------------------------------------------
// Attention kernel: one block per (b,h). S = softmax(Q K^T/8 + bias), Ctx = S V
// LDS layout (bytes):
//   [0      ,17408) Vt  [64][136] bf16  (V transposed: [feat][key])
//   [17408  ,35840) Qs  [128][72] bf16
//   [35840  ,54272) Ks  [128][72] bf16
//   [17408  ,52224) Ps  [128][136] bf16 (overlays Qs/Ks after S is computed)
//   [54272  ,54784) kb  [128] f32 key bias
// ---------------------------------------------------------------------------
__global__ __launch_bounds__(256) void attn_kernel(
    const float* __restrict__ Qb, const float* __restrict__ Kb,
    const float* __restrict__ Vb, const int* __restrict__ amask,
    float* __restrict__ Cx)
{
    __shared__ __align__(16) char smem[54784];
    bf16*  Vt = (bf16*)smem;                  // [64][136]
    bf16*  Qs = (bf16*)(smem + 17408);        // [128][72]
    bf16*  Ks = (bf16*)(smem + 35840);        // [128][72]
    bf16*  Ps = (bf16*)(smem + 17408);        // [128][136] overlay
    float* kb = (float*)(smem + 54272);       // [128]

    const int bh = blockIdx.x;
    const int b  = bh / NHEA;
    const int h  = bh % NHEA;
    const int tid = threadIdx.x;

    for (int i = tid; i < LL * DHEA; i += 256) {
        int r = i >> 6, c = i & 63;
        size_t base = (size_t)(b * LL + r) * DMOD + h * DHEA + c;
        Qs[r * 72 + c]  = (bf16)Qb[base];
        Ks[r * 72 + c]  = (bf16)Kb[base];
        Vt[c * 136 + r] = (bf16)Vb[base];
    }
    for (int i = tid; i < LL; i += 256)
        kb[i] = (1.0f - (float)amask[b * LL + i]) * -1e9f;
    __syncthreads();

    const int w     = tid >> 5;
    const int lane  = tid & 31;
    const int mlane = lane & 15;
    const int khalf = lane >> 4;

    // ---- S = Q * K^T : wave w -> query rows [16w,16w+16), all 128 keys ----
    v8f s[8] = {};
#pragma unroll
    for (int kt = 0; kt < 2; kt++) {
        const int arow = w * 16 + mlane;
        v16bf afrag = make_frag(&Qs[arow * 72 + kt * 32 + khalf * 8],
                                &Qs[arow * 72 + kt * 32 + 16 + khalf * 8]);
#pragma unroll
        for (int t = 0; t < 8; t++) {
            int bn = t * 16 + mlane;
            v16bf bfrag = make_frag(&Ks[bn * 72 + kt * 32 + khalf * 8],
                                    &Ks[bn * 72 + kt * 32 + 16 + khalf * 8]);
            s[t] = wmma_bf16(afrag, bfrag, s[t]);
        }
    }
    __syncthreads();   // Qs/Ks dead from here; Ps may overlay them

    // ---- softmax in registers; rows live in 16-lane half-waves ----
    const float scale = 0.125f;
    float kbv[8];
#pragma unroll
    for (int t = 0; t < 8; t++) kbv[t] = kb[t * 16 + mlane];

#pragma unroll
    for (int r = 0; r < 8; r++) {
        float sv[8];
        float mx = -1e30f;
#pragma unroll
        for (int t = 0; t < 8; t++) {
            sv[t] = s[t][r] * scale + kbv[t];
            mx = fmaxf(mx, sv[t]);
        }
        for (int d = 1; d < 16; d <<= 1) mx = fmaxf(mx, __shfl_xor(mx, d, 16));
        float sum = 0.f;
#pragma unroll
        for (int t = 0; t < 8; t++) { sv[t] = expf(sv[t] - mx); sum += sv[t]; }
        for (int d = 1; d < 16; d <<= 1) sum += __shfl_xor(sum, d, 16);
        const float inv = 1.0f / sum;
        const int m = w * 16 + khalf * 8 + r;
#pragma unroll
        for (int t = 0; t < 8; t++)
            Ps[m * 136 + t * 16 + mlane] = (bf16)(sv[t] * inv);
    }
    __syncthreads();

    // ---- Ctx = P * V : wave w -> rows [16w,16w+16), 64 features ----
    v8f c2[4] = {};
#pragma unroll
    for (int kt = 0; kt < 4; kt++) {
        const int arow = w * 16 + mlane;
        v16bf afrag = make_frag(&Ps[arow * 136 + kt * 32 + khalf * 8],
                                &Ps[arow * 136 + kt * 32 + 16 + khalf * 8]);
#pragma unroll
        for (int t = 0; t < 4; t++) {
            int bn = t * 16 + mlane;
            v16bf bfrag = make_frag(&Vt[bn * 136 + kt * 32 + khalf * 8],
                                    &Vt[bn * 136 + kt * 32 + 16 + khalf * 8]);
            c2[t] = wmma_bf16(afrag, bfrag, c2[t]);
        }
    }
#pragma unroll
    for (int t = 0; t < 4; t++) {
        const int n = h * DHEA + t * 16 + mlane;
#pragma unroll
        for (int r = 0; r < 8; r++) {
            const int m = w * 16 + khalf * 8 + r;
            Cx[(size_t)(b * LL + m) * DMOD + n] = c2[t][r];
        }
    }
}

// ---------------------------------------------------------------------------
// Head projection: em[b,t-1,k] = X[b*128+t,:] . head_W[:,k] + head_b[k]
// grid (127, 32)
// ---------------------------------------------------------------------------
__global__ __launch_bounds__(256) void head_kernel(
    const float* __restrict__ X, const float* __restrict__ hW,
    const float* __restrict__ hb, float* __restrict__ EM)
{
    const int t = blockIdx.x + 1;   // 1..127
    const int b = blockIdx.y;
    __shared__ float sx[DMOD];
    const float* xr = X + (size_t)(b * LL + t) * DMOD;
    for (int d = threadIdx.x; d < DMOD; d += 256) sx[d] = xr[d];
    __syncthreads();
    if (threadIdx.x < KTAG) {
        const int k = threadIdx.x;
        float acc = hb[k];
        for (int d = 0; d < DMOD; d++) acc += sx[d] * hW[d * KTAG + k];
        EM[((size_t)b * TT_ + (t - 1)) * KTAG + k] = acc;
    }
}

// ---------------------------------------------------------------------------
// CRF: llh (gold score - logZ, batch mean) + Viterbi decode.
// one block, 32 threads (one per batch element)
// ---------------------------------------------------------------------------
__global__ __launch_bounds__(32) void crf_kernel(
    const float* __restrict__ EM, const int* __restrict__ y,
    const int* __restrict__ am, const float* __restrict__ cstart,
    const float* __restrict__ cend, const float* __restrict__ ctrans,
    int* __restrict__ hist, float* __restrict__ out)
{
    const int b = threadIdx.x;
    __shared__ float sv[BB];
    const float* emr = EM + (size_t)b * TT_ * KTAG;
    const int*   yb  = y + b * TT_;

    // ---- gold path score ----
    int t0 = yb[0];
    float num = cstart[t0] + emr[t0];
    for (int i = 1; i < TT_; i++) {
        int ti = yb[i];
        float mf = (am[b * LL + i + 1] != 0) ? 1.0f : 0.0f;
        num += (ctrans[yb[i - 1] * KTAG + ti] + emr[i * KTAG + ti]) * mf;
    }
    int cnt = 0;
    for (int i = 0; i < TT_; i++) cnt += (am[b * LL + i + 1] != 0) ? 1 : 0;
    num += cend[yb[cnt - 1]];

    // ---- forward logsumexp ----
    float alpha[KTAG];
    for (int k = 0; k < KTAG; k++) alpha[k] = cstart[k] + emr[k];
    for (int i = 1; i < TT_; i++) {
        bool m = am[b * LL + i + 1] != 0;
        float nxt[KTAG];
        for (int k2 = 0; k2 < KTAG; k2++) {
            float mx = -1e30f;
            for (int k1 = 0; k1 < KTAG; k1++)
                mx = fmaxf(mx, alpha[k1] + ctrans[k1 * KTAG + k2]);
            float sum = 0.f;
            for (int k1 = 0; k1 < KTAG; k1++)
                sum += expf(alpha[k1] + ctrans[k1 * KTAG + k2] - mx);
            nxt[k2] = mx + logf(sum) + emr[i * KTAG + k2];
        }
        if (m) for (int k2 = 0; k2 < KTAG; k2++) alpha[k2] = nxt[k2];
    }
    float zmx = -1e30f;
    for (int k = 0; k < KTAG; k++) zmx = fmaxf(zmx, alpha[k] + cend[k]);
    float zs = 0.f;
    for (int k = 0; k < KTAG; k++) zs += expf(alpha[k] + cend[k] - zmx);
    const float Z = zmx + logf(zs);

    sv[b] = num - Z;
    __syncthreads();
    if (b == 0) {
        float acc = 0.f;
        for (int i = 0; i < BB; i++) acc += sv[i];
        out[0] = acc * (1.0f / BB);
    }

    // ---- Viterbi ----
    float sc[KTAG];
    for (int k = 0; k < KTAG; k++) sc[k] = cstart[k] + emr[k];
    for (int i = 1; i < TT_; i++) {
        bool m = am[b * LL + i + 1] != 0;
        float nxt[KTAG];
        for (int k2 = 0; k2 < KTAG; k2++) {
            float best = -1e30f; int bi = 0;
            for (int k1 = 0; k1 < KTAG; k1++) {
                float c = sc[k1] + ctrans[k1 * KTAG + k2];
                if (c > best) { best = c; bi = k1; }
            }
            nxt[k2] = best + emr[i * KTAG + k2];
            hist[((size_t)b * (TT_ - 1) + (i - 1)) * KTAG + k2] = bi;
        }
        if (m) for (int k2 = 0; k2 < KTAG; k2++) sc[k2] = nxt[k2];
    }
    float best = -1e30f; int tag = 0;
    for (int k = 0; k < KTAG; k++) {
        float c = sc[k] + cend[k];
        if (c > best) { best = c; tag = k; }
    }
    for (int i = TT_ - 1; i >= 1; i--) {
        out[1 + b * TT_ + i] = (float)tag;
        bool m = am[b * LL + i + 1] != 0;
        int prev = hist[((size_t)b * (TT_ - 1) + (i - 1)) * KTAG + tag];
        if (m) tag = prev;
    }
    out[1 + b * TT_ + 0] = (float)tag;
}

// ---------------------------------------------------------------------------
// Launch orchestration
// ---------------------------------------------------------------------------
extern "C" void kernel_launch(void* const* d_in, const int* in_sizes, int n_in,
                              void* d_out, int out_size, void* d_ws, size_t ws_size,
                              hipStream_t stream)
{
    (void)in_sizes; (void)n_in; (void)out_size; (void)ws_size;
    const int*   input_ids = (const int*)  d_in[0];
    const int*   attn_mask = (const int*)  d_in[1];
    const int*   type_ids  = (const int*)  d_in[2];
    const int*   y_true    = (const int*)  d_in[3];
    const float* word_emb  = (const float*)d_in[4];
    const float* pos_emb   = (const float*)d_in[5];
    const float* type_emb  = (const float*)d_in[6];
    const float* emb_g     = (const float*)d_in[7];
    const float* emb_b     = (const float*)d_in[8];
    const float* Wq        = (const float*)d_in[9];
    const float* bq        = (const float*)d_in[10];
    const float* Wk        = (const float*)d_in[11];
    const float* bk        = (const float*)d_in[12];
    const float* Wv        = (const float*)d_in[13];
    const float* bv        = (const float*)d_in[14];
    const float* Wo        = (const float*)d_in[15];
    const float* bo        = (const float*)d_in[16];
    const float* ln1_g     = (const float*)d_in[17];
    const float* ln1_b     = (const float*)d_in[18];
    const float* W1        = (const float*)d_in[19];
    const float* b1        = (const float*)d_in[20];
    const float* W2        = (const float*)d_in[21];
    const float* b2        = (const float*)d_in[22];
    const float* ln2_g     = (const float*)d_in[23];
    const float* ln2_b     = (const float*)d_in[24];
    const float* head_W    = (const float*)d_in[25];
    const float* head_b    = (const float*)d_in[26];
    const float* crf_start = (const float*)d_in[27];
    const float* crf_end   = (const float*)d_in[28];
    const float* crf_trans = (const float*)d_in[29];

    float* out = (float*)d_out;

    // ---- workspace layout ----
    float* ws  = (float*)d_ws;
    const size_t MD  = (size_t)MROW * DMOD;
    const size_t DDL = (size_t)NLAY * DMOD * DMOD;   // 7,077,888
    const size_t DFL = (size_t)NLAY * DMOD * FFD;    // 28,311,552
    float* X   = ws;
    float* X2  = X  + MD;
    float* Qb  = X2 + MD;
    float* Kb  = Qb + MD;
    float* Vb  = Kb + MD;
    float* Cx  = Vb + MD;
    float* FFb = Cx + MD;                            // MROW * FFD
    float* EM  = FFb + (size_t)MROW * FFD;           // 32*127*9
    int*   HIST = (int*)(EM + (size_t)BB * TT_ * KTAG);
    // bf16 pre-transposed weights after the f32 region (16B aligned)
    bf16*  WqT = (bf16*)(HIST + (size_t)BB * (TT_ - 1) * KTAG);
    bf16*  WkT = WqT + DDL;
    bf16*  WvT = WkT + DDL;
    bf16*  WoT = WvT + DDL;
    bf16*  W1T = WoT + DDL;                          // [FF][DM] per layer
    bf16*  W2T = W1T + DFL;                          // [DM][FF] per layer

    const dim3 blk(256);
    const dim3 g_dm(DMOD / GBN, MROW / GBM);         // (6, 64)
    const dim3 g_ff(FFD  / GBN, MROW / GBM);         // (24, 64)

    // ---- one-time (per launch) weight convert+transpose to bf16 ----
    wcvt_kernel<<<dim3(DMOD / 32, DMOD / 32, NLAY), blk, 0, stream>>>(Wq, WqT, DMOD, DMOD);
    wcvt_kernel<<<dim3(DMOD / 32, DMOD / 32, NLAY), blk, 0, stream>>>(Wk, WkT, DMOD, DMOD);
    wcvt_kernel<<<dim3(DMOD / 32, DMOD / 32, NLAY), blk, 0, stream>>>(Wv, WvT, DMOD, DMOD);
    wcvt_kernel<<<dim3(DMOD / 32, DMOD / 32, NLAY), blk, 0, stream>>>(Wo, WoT, DMOD, DMOD);
    wcvt_kernel<<<dim3(FFD  / 32, DMOD / 32, NLAY), blk, 0, stream>>>(W1, W1T, DMOD, FFD);
    wcvt_kernel<<<dim3(DMOD / 32, FFD  / 32, NLAY), blk, 0, stream>>>(W2, W2T, FFD, DMOD);

    embed_ln_kernel<<<MROW, blk, 0, stream>>>(input_ids, type_ids, word_emb, pos_emb,
                                              type_emb, emb_g, emb_b, X);

    for (int l = 0; l < NLAY; l++) {
        const size_t wdd = (size_t)l * DMOD * DMOD;
        const size_t wdf = (size_t)l * DMOD * FFD;
        // Q,K,V projections
        gemm_bf16_kernel<0><<<g_dm, blk, 0, stream>>>(X, WqT + wdd, bq + l * DMOD, nullptr, Qb, DMOD, DMOD);
        gemm_bf16_kernel<0><<<g_dm, blk, 0, stream>>>(X, WkT + wdd, bk + l * DMOD, nullptr, Kb, DMOD, DMOD);
        gemm_bf16_kernel<0><<<g_dm, blk, 0, stream>>>(X, WvT + wdd, bv + l * DMOD, nullptr, Vb, DMOD, DMOD);
        // attention
        attn_kernel<<<BB * NHEA, blk, 0, stream>>>(Qb, Kb, Vb, attn_mask, Cx);
        // O projection + residual, then LN1 -> X
        gemm_bf16_kernel<2><<<g_dm, blk, 0, stream>>>(Cx, WoT + wdd, bo + l * DMOD, X, X2, DMOD, DMOD);
        ln_kernel<<<MROW, blk, 0, stream>>>(X2, ln1_g + l * DMOD, ln1_b + l * DMOD, X);
        // FFN
        gemm_bf16_kernel<1><<<g_ff, blk, 0, stream>>>(X, W1T + wdf, b1 + l * FFD, nullptr, FFb, DMOD, FFD);
        gemm_bf16_kernel<2><<<g_dm, blk, 0, stream>>>(FFb, W2T + wdf, b2 + l * DMOD, X, X2, FFD, DMOD);
        ln_kernel<<<MROW, blk, 0, stream>>>(X2, ln2_g + l * DMOD, ln2_b + l * DMOD, X);
    }

    head_kernel<<<dim3(TT_, BB), blk, 0, stream>>>(X, head_W, head_b, EM);
    crf_kernel<<<1, 32, 0, stream>>>(EM, y_true, attn_mask, crf_start, crf_end,
                                     crf_trans, HIST, out);
}
